// XLSTM_46385646797340
// MI455X (gfx1250) — matrix operand order repaired and checked by reading, not compiled, not run on hardware
//
#include <hip/hip_runtime.h>
#include <hip/hip_bf16.h>

typedef __bf16 bf16;
typedef __attribute__((ext_vector_type(16))) __bf16 v16bf;
typedef __attribute__((ext_vector_type(8)))  float  v8f;
typedef int v4i __attribute__((vector_size(16)));
typedef __attribute__((address_space(1))) v4i* g4ptr;   // global
typedef __attribute__((address_space(3))) v4i* l4ptr;   // LDS

#define DEV __device__ __forceinline__

#if __has_builtin(__builtin_amdgcn_global_load_async_to_lds_b128) && \
    __has_builtin(__builtin_amdgcn_s_wait_asynccnt)
#define HAVE_ASYNC 1
#else
#define HAVE_ASYNC 0
#endif

// ---------------- model dims ----------------
constexpr int T_   = 512;
constexpr int Bb   = 4;
constexpr int Dd   = 512;
constexpr int Hh   = 8;
constexpr int UPM  = 1024;
constexpr int UPS  = 682;   // sLSTM up half-width
constexpr int UPS2 = 1364;  // 2*UPS
constexpr int UPS2P= 1376;  // padded to 16
constexpr int UPSP = 704;   // padded K for sLSTM down (multiple of 32)

// ---------------- grid ----------------
constexpr int NWG = 16, BLK = 256, WPG = BLK / 32, GW = NWG * WPG; // 128 waves

// ---------------- input tensor indices (sorted-pytree flatten) ----------------
constexpr int LIN[4] = {3, 28, 48, 73};
enum { M_I_WF_B=0,M_I_WF_W,M_I_WI_B,M_I_WI_W,M_I_WK_B,M_I_WK_W,M_I_WO_B,M_I_WO_W,
       M_I_WQ_B,M_I_WQ_W,M_I_WV_B,M_I_WV_W,M_I_CONV_B,M_I_CONV_W,M_I_DN_B,M_I_DN_W,
       M_I_GN_B,M_I_GN_G,M_I_LN_B,M_I_LN_G,M_I_SKIP_W,M_I_UPL_B,M_I_UPL_W,M_I_UPR_B,M_I_UPR_W };
enum { S_I_RF=0,S_I_RI,S_I_RO,S_I_RZ,S_I_WF_B,S_I_WF_W,S_I_WI_B,S_I_WI_W,
       S_I_WO_B,S_I_WO_W,S_I_WZ_B,S_I_WZ_W,S_I_DN_B,S_I_DN_W,S_I_GN_B,S_I_GN_G,
       S_I_LN_B,S_I_LN_G,S_I_UP_B,S_I_UP_W };

// ---------------- bf16 weight workspace layout (elements) ----------------
constexpr size_t M_UPL=0, M_UPR=524288, M_WQ=786432, M_WK=1310720, M_WV=1835008,
                 M_WO=2359296, M_WI=2883584, M_WF=2899968, M_SKIP=2916352,
                 M_DOWN=3440640, M_TOT=3702784;
constexpr size_t S_WZ=0, S_WI=262144, S_WF=524288, S_WO=786432,
                 S_RZ=1048576, S_RI=1081344, S_RF=1114112, S_RO=1146880,
                 S_UP=1179648, S_DOWN=1884160, S_TOT=2244608;
constexpr size_t LBF[4] = {0, M_TOT, M_TOT+S_TOT, 2*M_TOT+S_TOT};
constexpr size_t HEADW_BF = 2*M_TOT + 2*S_TOT;
constexpr size_t BF_TOTAL = HEADW_BF + 512*512;
constexpr size_t WS_F32B  = 256 + BF_TOTAL*2;              // 24314112 (256B aligned)

// ---------------- fp32 scratch layout (float offsets inside F32 region) ----------------
constexpr size_t F_X=0, F_XT=2048, F_RT=6144, F_Q=8192, F_K=10240, F_V=12288,
                 F_OP=14336, F_SK=16384, F_IF=18432, F_NUM=18496, F_DEN=20544,
                 F_UP=22656, F_GATES=30976;
constexpr size_t F_CP16=39168;                 // 4096 bf16 staging (2048 floats)
constexpr size_t F_STATE0=41216;
constexpr size_t SM_C[2]={41216,182592}, SM_N[2]={172288,313664}, SM_M[2]={174336,315712};
constexpr size_t SS_C[2]={174400,315776}, SS_N[2]={176448,317824}, SS_M[2]={180544,321920};
constexpr size_t F_HS16[2]={323968,324992};    // h_{t-1} bf16, per sLSTM slot (zeroed)
constexpr size_t F_STATE_END=326016;
constexpr size_t F_OUTS=326016;
constexpr size_t WS_NEEDED = WS_F32B + (F_OUTS + (size_t)T_*Bb*Dd)*4;

struct Params { const float* p[96]; };

// ---------------- helpers ----------------
DEV float sigm(float x) { return 1.f / (1.f + __expf(-x)); }
DEV float wredsum(float v) {
  #pragma unroll
  for (int o = 16; o; o >>= 1) v += __shfl_down(v, o, 32);
  return v;
}
DEV v8f zero8() { v8f a = {0.f,0.f,0.f,0.f,0.f,0.f,0.f,0.f}; return a; }

DEV v16bf ldBfrag(const bf16* wrow, int k0) {
  union { v16bf v; float4 q[2]; } u;
  const float4* p = (const float4*)(wrow + k0);
  u.q[0] = p[0]; u.q[1] = p[2];
  return u.v;
}
DEV v16bf ldAfrag(const bf16* arow, int k0) {
  union { v16bf v; float4 q[2]; } u;
  u.q[0] = *(const float4*)(arow + k0);
  u.q[1] = *(const float4*)(arow + k0 + 16);
  return u.v;
}

// y(16x16 tile, rows 0..3 valid) += A(4xK bf16 LDS; rows>=4 -> zrow) * W(NxK)^T
// software-pipelined: fragment loads for s+1 overlap the WMMA for s
DEV v8f wmma_kloop(const bf16* A, int ldA, const bf16* W, int ldW,
                   int ksteps, int n0, int lane, v8f acc, const bf16* zrow) {
  const int m = lane & 15, half = lane >> 4;
  const bf16* arow = (m < Bb) ? (A + m * ldA + 8 * half) : (zrow + 8 * half);
  const bf16* wrow = W + (size_t)(n0 + m) * ldW + 8 * half;
  v16bf a = ldAfrag(arow, 0);
  v16bf b = ldBfrag(wrow, 0);
  for (int s = 0; s < ksteps; ++s) {
    int sn = (s + 1 < ksteps) ? s + 1 : s;
    if ((s & 7) == 0) __builtin_prefetch(wrow + s * 32 + 256, 0, 1);
    v16bf an = ldAfrag(arow, sn * 32);
    v16bf bn = ldBfrag(wrow, sn * 32);
    acc = __builtin_amdgcn_wmma_f32_16x16x32_bf16(false, a, false, b,
                                                  (short)0, acc, false, false);
    a = an; b = bn;
  }
  return acc;
}

DEV void store4(v8f acc, const float* bias, int n0, float* out, int ldo,
                int lane, float scale) {
  if (lane < 16) {
    int n = n0 + lane;
    float bb = bias ? bias[n] : 0.f;
    #pragma unroll
    for (int r = 0; r < Bb; ++r) out[r * ldo + n] = (acc[r] + bb) * scale;
  }
}
// store fp32 + bf16 staging copy
DEV void store4b(v8f acc, const float* bias, int n0, float* out, int ldo,
                 bf16* out16, int ld16, int lane) {
  if (lane < 16) {
    int n = n0 + lane;
    float bb = bias ? bias[n] : 0.f;
    #pragma unroll
    for (int r = 0; r < Bb; ++r) {
      float v = acc[r] + bb;
      out[r * ldo + n] = v;
      out16[r * ld16 + n] = (bf16)v;
    }
  }
}

// global bf16 -> LDS staging (async DMA when available)
DEV void lds_fill_issue(bf16* ldst, const bf16* gsrc, int bytes, int tid) {
#if HAVE_ASYNC
  for (int off = tid * 16; off < bytes; off += BLK * 16)
    __builtin_amdgcn_global_load_async_to_lds_b128(
        (g4ptr)((char*)gsrc + off),
        (l4ptr)((char*)ldst + off), 0, 0);
#else
  for (int off = tid * 16; off < bytes; off += BLK * 16)
    *(float4*)((char*)ldst + off) = *(const float4*)((const char*)gsrc + off);
#endif
}
DEV void lds_fill_wait() {
#if HAVE_ASYNC
  __builtin_amdgcn_s_wait_asynccnt(0);
#endif
}

// monotonic-counter grid barrier
DEV void gridbar(unsigned* cnt) {
  __syncthreads();
  if (threadIdx.x == 0) {
    __threadfence();
    unsigned a = __hip_atomic_fetch_add(cnt, 1u, __ATOMIC_ACQ_REL, __HIP_MEMORY_SCOPE_AGENT);
    unsigned target = (a / NWG + 1u) * NWG;
    while (__hip_atomic_load(cnt, __ATOMIC_ACQUIRE, __HIP_MEMORY_SCOPE_AGENT) < target)
      __builtin_amdgcn_s_sleep(2);
  }
  __syncthreads();
}

// ---------------- init: zero barrier + recurrent states ----------------
__global__ void init_kernel(unsigned char* ws) {
  int i = blockIdx.x * 256 + threadIdx.x;
  if (i < 64) ((unsigned*)ws)[i] = 0u;
  float* F = (float*)(ws + WS_F32B);
  if (i < (int)(F_STATE_END - F_STATE0)) F[F_STATE0 + i] = 0.f;
}

// ---------------- weight fp32 -> padded bf16 ----------------
__global__ void conv_bf16_kernel(const float* src, bf16* dst, int N, int K, int pN, int pK) {
  int i = blockIdx.x * 256 + threadIdx.x;
  if (i >= pN * pK) return;
  int n = i / pK, k = i % pK;
  float v = (n < N && k < K) ? src[(size_t)n * K + k] : 0.f;
  dst[i] = (bf16)v;
}

// ---------------- persistent xLSTM scan ----------------
__global__ __launch_bounds__(BLK) void xlstm_kernel(Params prm, unsigned char* ws) {
  unsigned* bar = (unsigned*)ws;
  bf16* wbf = (bf16*)(ws + 256);
  float* F = (float*)(ws + WS_F32B);
  __shared__ __align__(16) bf16 sA[4 * 1024];
  __shared__ __align__(16) bf16 sB[4 * 1024];
  __shared__ __align__(16) bf16 szero[1024];
  __shared__ float sred[16];

  const int tid = threadIdx.x, lane = tid & 31, wid = tid >> 5;
  const int gw = blockIdx.x * WPG + wid;
  const float* xin = prm.p[0];
  float* X = F + F_X;
  bf16* CP16 = (bf16*)(F + F_CP16);

  for (int i = tid; i < 1024; i += BLK) szero[i] = (bf16)0.f;
  __syncthreads();

  for (int t = 0; t < T_; ++t) {
    for (int L = 0; L < 4; ++L) {
      const int base = LIN[L];
      bf16* WB = wbf + LBF[L];
      const int slot = L >> 1;
      auto inrow = [&](int b) -> const float* {
        return (L == 0) ? (xin + (size_t)b * T_ * Dd + (size_t)t * Dd) : (X + b * Dd);
      };
      // -------- layer_norm into sA (redundant per WG, avoids a grid sync) --------
      const float* lng = prm.p[base + ((L & 1) ? S_I_LN_G : M_I_LN_G)];
      const float* lnb = prm.p[base + ((L & 1) ? S_I_LN_B : M_I_LN_B)];
      if (wid < Bb) {
        const float* xr = inrow(wid);
        float s = 0.f, s2 = 0.f;
        for (int j = lane; j < Dd; j += 32) { float v = xr[j]; s += v; s2 += v * v; }
        s = wredsum(s); s2 = wredsum(s2);
        if (lane == 0) {
          float mu = s / Dd;
          sred[wid * 2] = mu;
          sred[wid * 2 + 1] = rsqrtf(s2 / Dd - mu * mu + 1e-5f);
        }
      }
      __syncthreads();
      for (int i = tid; i < Bb * Dd; i += BLK) {
        int b = i >> 9, j = i & 511;
        sA[i] = (bf16)((inrow(b)[j] - sred[b * 2]) * sred[b * 2 + 1] * lng[j] + lnb[j]);
      }

      if ((L & 1) == 0) {
        // ================= mLSTM =================
        float *XT = F + F_XT, *RT = F + F_RT, *Q = F + F_Q, *Kb = F + F_K, *V = F + F_V;
        float *OP = F + F_OP, *SK = F + F_SK, *IF = F + F_IF, *NUM = F + F_NUM;
        float *DEN = F + F_DEN;
        float *Cst = F + SM_C[slot], *Nst = F + SM_N[slot], *Mpp = F + SM_M[slot];
        __syncthreads();
        // Stage A: up_l (fp32 + bf16 staging) / up_r
        for (int tile = gw; tile < 96; tile += GW) {
          v8f acc = zero8();
          if (tile < 64) {
            int n0 = tile * 16;
            acc = wmma_kloop(sA, Dd, WB + M_UPL, Dd, 16, n0, lane, acc, szero);
            store4b(acc, prm.p[base + M_I_UPL_B], n0, XT, UPM, CP16, UPM, lane);
          } else {
            int n0 = (tile - 64) * 16;
            acc = wmma_kloop(sA, Dd, WB + M_UPR, Dd, 16, n0, lane, acc, szero);
            store4(acc, prm.p[base + M_I_UPR_B], n0, RT, Dd, lane, 1.f);
          }
        }
        gridbar(bar);
        // Stage B: async x_t -> sB overlapping conv+silu -> sA; then 6 GEMM groups
        lds_fill_issue(sB, CP16, Bb * UPM * 2, tid);
        {
          const float cb = prm.p[base + M_I_CONV_B][0];
          const float* cw = prm.p[base + M_I_CONV_W];
          for (int i = tid; i < Bb * UPM; i += BLK) {
            int b = i >> 10, j = i & 1023;
            float a = cb;
            #pragma unroll
            for (int k = 0; k < 4; ++k) {
              int jj = j - 3 + k;
              if (jj >= 0) a += cw[k] * XT[b * UPM + jj];
            }
            sA[i] = (bf16)(a * sigm(a));        // x_c = silu(conv)
          }
        }
        lds_fill_wait();
        __syncthreads();
        for (int tile = gw; tile < 162; tile += GW) {
          v8f acc = zero8();
          if (tile < 32) {
            int n0 = tile * 16;
            acc = wmma_kloop(sA, UPM, WB + M_WQ, UPM, 32, n0, lane, acc, szero);
            store4(acc, prm.p[base + M_I_WQ_B], n0, Q, Dd, lane, 1.f);
          } else if (tile < 64) {
            int n0 = (tile - 32) * 16;
            acc = wmma_kloop(sA, UPM, WB + M_WK, UPM, 32, n0, lane, acc, szero);
            store4(acc, prm.p[base + M_I_WK_B], n0, Kb, Dd, lane, 0.125f);
          } else if (tile < 96) {
            int n0 = (tile - 64) * 16;
            acc = wmma_kloop(sB, UPM, WB + M_WV, UPM, 32, n0, lane, acc, szero);
            store4(acc, prm.p[base + M_I_WV_B], n0, V, Dd, lane, 1.f);
          } else if (tile < 128) {
            int n0 = (tile - 96) * 16;
            acc = wmma_kloop(sB, UPM, WB + M_WO, UPM, 32, n0, lane, acc, szero);
            store4(acc, prm.p[base + M_I_WO_B], n0, OP, Dd, lane, 1.f);
          } else if (tile < 160) {
            int n0 = (tile - 128) * 16;
            acc = wmma_kloop(sA, UPM, WB + M_SKIP, UPM, 32, n0, lane, acc, szero);
            store4(acc, (const float*)nullptr, n0, SK, Dd, lane, 1.f);
          } else {
            bool isF = (tile == 161);
            acc = wmma_kloop(sA, UPM, WB + (isF ? M_WF : M_WI), UPM, 32, 0, lane, acc, szero);
            if (lane < Hh) {
              const float* bia = prm.p[base + (isF ? M_I_WF_B : M_I_WI_B)];
              #pragma unroll
              for (int r = 0; r < Bb; ++r) IF[(isF ? 32 : 0) + r * 8 + lane] = acc[r] + bia[lane];
            }
          }
        }
        gridbar(bar);
        // Stage C: gates, C/n state update, num/den reductions
        {
          int par = t & 1;
          for (int r = gw; r < 2048; r += GW) {
            int b = r >> 9, h = (r >> 6) & 7, d = r & 63;
            float iv = IF[b * 8 + h], fv = IF[32 + b * 8 + h], mo = Mpp[par * 32 + b * 8 + h];
            float mt = fmaxf(fv + mo, iv), ie = __expf(iv - mt), fe = __expf(fv + mo - mt);
            float vv = V[b * Dd + h * 64 + d];
            int pb = b * Dd + h * 64;
            size_t cb_ = (size_t)r * 64;
            float c0 = fe * Cst[cb_ + lane]      + ie * vv * Kb[pb + lane];
            float c1 = fe * Cst[cb_ + lane + 32] + ie * vv * Kb[pb + lane + 32];
            Cst[cb_ + lane] = c0; Cst[cb_ + lane + 32] = c1;
            float s = wredsum(c0 * Q[pb + lane] + c1 * Q[pb + lane + 32]);
            if (lane == 0) NUM[r] = s;
          }
          for (int r2 = gw; r2 < 32; r2 += GW) {
            int b = r2 >> 3, h = r2 & 7;
            float iv = IF[r2], fv = IF[32 + r2], mo = Mpp[par * 32 + r2];
            float mt = fmaxf(fv + mo, iv), ie = __expf(iv - mt), fe = __expf(fv + mo - mt);
            int pb = b * Dd + h * 64;
            float n0v = fe * Nst[r2 * 64 + lane]      + ie * Kb[pb + lane];
            float n1v = fe * Nst[r2 * 64 + lane + 32] + ie * Kb[pb + lane + 32];
            Nst[r2 * 64 + lane] = n0v; Nst[r2 * 64 + lane + 32] = n1v;
            float s = wredsum(n0v * Q[pb + lane] + n1v * Q[pb + lane + 32]);
            if (lane == 0) { DEN[r2] = fmaxf(fabsf(s), 1.f); Mpp[(par ^ 1) * 32 + r2] = mt; }
          }
        }
        gridbar(bar);
        // Stage D: h_t, group_norm, +skip, *silu(r_t) -> bf16 staging
        {
          const float* gng = prm.p[base + M_I_GN_G];
          const float* gnb = prm.p[base + M_I_GN_B];
          for (int r2 = gw; r2 < 32; r2 += GW) {
            int b = r2 >> 3, h = r2 & 7, pb = b * Dd + h * 64;
            float dn = DEN[r2];
            float h0 = sigm(OP[pb + lane])      * NUM[(r2 << 6) + lane]      / dn;
            float h1 = sigm(OP[pb + lane + 32]) * NUM[(r2 << 6) + lane + 32] / dn;
            float s  = wredsum(h0 + h1);
            float s2 = wredsum(h0 * h0 + h1 * h1);
            float mu = __shfl(s, 0, 32) / 64.f;
            float rs = rsqrtf(__shfl(s2, 0, 32) / 64.f - mu * mu + 1e-5f);
            #pragma unroll
            for (int e = 0; e < 2; ++e) {
              int d = lane + 32 * e, j = h * 64 + d, x = b * Dd + j;
              float hv = e ? h1 : h0;
              float val = (hv - mu) * rs * gng[j] + gnb[j] + SK[x];
              float rv = RT[x];
              CP16[x] = (bf16)(val * (rv * sigm(rv)));
            }
          }
        }
        gridbar(bar);
        // Stage E: async staging -> sA, down GEMM + residual
        lds_fill_issue(sA, CP16, Bb * Dd * 2, tid);
        lds_fill_wait();
        __syncthreads();
        {
          const float* dnb = prm.p[base + M_I_DN_B];
          for (int tile = gw; tile < 32; tile += GW) {
            int n0 = tile * 16;
            v8f acc = wmma_kloop(sA, Dd, WB + M_DOWN, Dd, 16, n0, lane, zero8(), szero);
            if (lane < 16) {
              int n = n0 + lane;
              float bb = dnb[n];
              #pragma unroll
              for (int r = 0; r < Bb; ++r) X[r * Dd + n] = acc[r] + bb + inrow(r)[n];
            }
          }
        }
        gridbar(bar);
      } else {
        // ================= sLSTM =================
        float *GT = F + F_GATES, *UP = F + F_UP;
        float *Cs = F + SS_C[slot], *Ns = F + SS_N[slot], *Ms = F + SS_M[slot];
        bf16* HSb = (bf16*)(F + F_HS16[slot]);
        lds_fill_issue(sB, HSb, Bb * Dd * 2, tid);   // h_{t-1} bf16 -> LDS
        lds_fill_wait();
        __syncthreads();
        // Stage A: 4 gate GEMMs with fused block-diag recurrence
        {
          const size_t woff[4] = {S_WZ, S_WI, S_WF, S_WO};
          const size_t roff[4] = {S_RZ, S_RI, S_RF, S_RO};
          const int bidx[4] = {S_I_WZ_B, S_I_WI_B, S_I_WF_B, S_I_WO_B};
          for (int tile = gw; tile < 128; tile += GW) {
            int g = tile >> 5, n0 = (tile & 31) * 16, hh = n0 >> 6;
            v8f acc = zero8();
            acc = wmma_kloop(sA, Dd, WB + woff[g], Dd, 16, n0, lane, acc, szero);
            acc = wmma_kloop(sB + hh * 64, Dd, WB + roff[g] + (size_t)hh * 4096, 64, 2,
                             n0 & 63, lane, acc, szero);
            store4(acc, prm.p[base + bidx[g]], n0, GT + g * 2048, Dd, lane, 1.f);
          }
        }
        gridbar(bar);
        // Stage B: gates + state update + group_norm -> staging (bf16)
        {
          const float* gng = prm.p[base + S_I_GN_G];
          const float* gnb = prm.p[base + S_I_GN_B];
          for (int r2 = gw; r2 < 32; r2 += GW) {
            int b = r2 >> 3, h = r2 & 7;
            float hv[2];
            #pragma unroll
            for (int e = 0; e < 2; ++e) {
              int d = lane + 32 * e, x = b * Dd + h * 64 + d;
              float z  = tanhf(GT[x]);
              float iv = GT[2048 + x];
              float fv = GT[4096 + x];
              float ov = sigm(GT[6144 + x]);
              float mo = Ms[x];
              float mt = fmaxf(fv + mo, iv), ie = __expf(iv - mt), fe = __expf(fv + mo - mt);
              float ct = fe * Cs[x] + ie * z;
              float nt = fe * Ns[x] + ie;
              float ht = ov * (ct / nt);
              Cs[x] = ct; Ns[x] = nt; Ms[x] = mt;
              HSb[x] = (bf16)ht;                 // next-step recurrence input
              hv[e] = ht;
            }
            float s  = wredsum(hv[0] + hv[1]);
            float s2 = wredsum(hv[0] * hv[0] + hv[1] * hv[1]);
            float mu = __shfl(s, 0, 32) / 64.f;
            float rs = rsqrtf(__shfl(s2, 0, 32) / 64.f - mu * mu + 1e-5f);
            #pragma unroll
            for (int e = 0; e < 2; ++e) {
              int d = lane + 32 * e, j = h * 64 + d, x = b * Dd + j;
              CP16[x] = (bf16)((hv[e] - mu) * rs * gng[j] + gnb[j]);
            }
          }
        }
        gridbar(bar);
        // Stage C: up GEMM (N padded 1364->1376)
        lds_fill_issue(sA, CP16, Bb * Dd * 2, tid);
        lds_fill_wait();
        __syncthreads();
        for (int tile = gw; tile < 86; tile += GW) {
          int n0 = tile * 16;
          v8f acc = wmma_kloop(sA, Dd, WB + S_UP, Dd, 16, n0, lane, zero8(), szero);
          if (lane < 16) {
            int n = n0 + lane;
            float bb = (n < UPS2) ? prm.p[base + S_I_UP_B][n] : 0.f;
            #pragma unroll
            for (int r = 0; r < Bb; ++r) UP[r * UPS2P + n] = acc[r] + bb;
          }
        }
        gridbar(bar);
        // Stage D: o1 + gelu(o2) -> staging bf16 (zero-pad K to 704)
        for (int i = blockIdx.x * BLK + tid; i < Bb * UPSP; i += NWG * BLK) {
          int b = i / UPSP, j = i % UPSP;
          float val = 0.f;
          if (j < UPS) {
            float o1 = UP[b * UPS2P + j], o2 = UP[b * UPS2P + UPS + j];
            val = o1 + 0.5f * o2 * (1.f + erff(o2 * 0.70710678f));
          }
          CP16[b * UPSP + j] = (bf16)val;
        }
        gridbar(bar);
        // Stage E: down GEMM (K=704) + residual (+outs on last layer)
        lds_fill_issue(sA, CP16, Bb * UPSP * 2, tid);
        lds_fill_wait();
        __syncthreads();
        {
          const float* dnb = prm.p[base + S_I_DN_B];
          float* OUTS = F + F_OUTS;
          for (int tile = gw; tile < 32; tile += GW) {
            int n0 = tile * 16;
            v8f acc = wmma_kloop(sA, UPSP, WB + S_DOWN, UPSP, UPSP / 32, n0, lane,
                                 zero8(), szero);
            if (lane < 16) {
              int n = n0 + lane;
              float bb = dnb[n];
              #pragma unroll
              for (int r = 0; r < Bb; ++r) {
                float val = acc[r] + bb + inrow(r)[n];
                X[r * Dd + n] = val;
                if (L == 3) OUTS[((size_t)t * Bb + r) * Dd + n] = val;
              }
            }
          }
        }
        gridbar(bar);
      }
    }
  }
}

// ---------------- vocabulary head: (T*B,512) @ (512,512)^T, full WMMA tiles ----------------
__global__ __launch_bounds__(BLK) void head_kernel(const float* outs, const bf16* hw,
                                                   const float* hb, float* out) {
  int lane = threadIdx.x & 31, wid = threadIdx.x >> 5;
  int gwv = blockIdx.x * WPG + wid, nw = gridDim.x * WPG;
  const int m = lane & 15, half = lane >> 4;
  for (int tile = gwv; tile < 128 * 32; tile += nw) {
    int tm = tile >> 5, tn = tile & 31;
    const float* arow = outs + (size_t)(tm * 16 + m) * Dd + 8 * half;
    const bf16*  wrow = hw + (size_t)(tn * 16 + m) * Dd + 8 * half;
    v8f acc = zero8();
    for (int s = 0; s < 16; ++s) {
      __builtin_prefetch(wrow + s * 32 + 128, 0, 1);
      v16bf a;
      #pragma unroll
      for (int e = 0; e < 8; ++e) a[e] = (bf16)arow[s * 32 + e];
      #pragma unroll
      for (int e = 0; e < 8; ++e) a[8 + e] = (bf16)arow[s * 32 + 16 + e];
      v16bf b = ldBfrag(wrow, s * 32);
      acc = __builtin_amdgcn_wmma_f32_16x16x32_bf16(false, a, false, b,
                                                    (short)0, acc, false, false);
    }
    int n = tn * 16 + (lane & 15);
    float bb = hb[n];
    int rbase = (lane >> 4) * 8;
    #pragma unroll
    for (int r = 0; r < 8; ++r) {
      int R = tm * 16 + rbase + r;
      int tt = R >> 2, b = R & 3;
      out[((size_t)b * T_ + tt) * Dd + n] = acc[r] + bb;   // (B,T,D)
    }
  }
}

// ---------------- launch ----------------
extern "C" void kernel_launch(void* const* d_in, const int* in_sizes, int n_in,
                              void* d_out, int out_size, void* d_ws, size_t ws_size,
                              hipStream_t stream) {
  if (ws_size < WS_NEEDED) return;
  unsigned char* ws = (unsigned char*)d_ws;
  bf16* wbf = (bf16*)(ws + 256);

  int initN = (int)(F_STATE_END - F_STATE0);
  init_kernel<<<(initN + 255) / 256, 256, 0, stream>>>(ws);

  struct CE { int src; size_t dst; int N, K, pN, pK; };
  CE tab[48]; int nt = 0;
  for (int li = 0; li < 4; li += 2) {          // mLSTM layers
    int bi = LIN[li]; size_t bo = LBF[li];
    tab[nt++] = {bi + M_I_UPL_W, bo + M_UPL, 1024, 512, 1024, 512};
    tab[nt++] = {bi + M_I_UPR_W, bo + M_UPR,  512, 512,  512, 512};
    tab[nt++] = {bi + M_I_WQ_W,  bo + M_WQ,   512,1024,  512,1024};
    tab[nt++] = {bi + M_I_WK_W,  bo + M_WK,   512,1024,  512,1024};
    tab[nt++] = {bi + M_I_WV_W,  bo + M_WV,   512,1024,  512,1024};
    tab[nt++] = {bi + M_I_WO_W,  bo + M_WO,   512,1024,  512,1024};
    tab[nt++] = {bi + M_I_WI_W,  bo + M_WI,     8,1024,   16,1024};
    tab[nt++] = {bi + M_I_WF_W,  bo + M_WF,     8,1024,   16,1024};
    tab[nt++] = {bi + M_I_SKIP_W,bo + M_SKIP, 512,1024,  512,1024};
    tab[nt++] = {bi + M_I_DN_W,  bo + M_DOWN, 512, 512,  512, 512};
  }
  for (int li = 1; li < 4; li += 2) {          // sLSTM layers
    int bi = LIN[li]; size_t bo = LBF[li];
    tab[nt++] = {bi + S_I_WZ_W, bo + S_WZ,   512, 512, 512, 512};
    tab[nt++] = {bi + S_I_WI_W, bo + S_WI,   512, 512, 512, 512};
    tab[nt++] = {bi + S_I_WF_W, bo + S_WF,   512, 512, 512, 512};
    tab[nt++] = {bi + S_I_WO_W, bo + S_WO,   512, 512, 512, 512};
    tab[nt++] = {bi + S_I_RZ,   bo + S_RZ,   512,  64, 512,  64};
    tab[nt++] = {bi + S_I_RI,   bo + S_RI,   512,  64, 512,  64};
    tab[nt++] = {bi + S_I_RF,   bo + S_RF,   512,  64, 512,  64};
    tab[nt++] = {bi + S_I_RO,   bo + S_RO,   512,  64, 512,  64};
    tab[nt++] = {bi + S_I_UP_W, bo + S_UP,  1364, 512,1376, 512};
    tab[nt++] = {bi + S_I_DN_W, bo + S_DOWN, 512, 682, 512, 704};
  }
  tab[nt++] = {2, HEADW_BF, 512, 512, 512, 512};   // head.w
  for (int i = 0; i < nt; ++i) {
    int tot = tab[i].pN * tab[i].pK;
    conv_bf16_kernel<<<(tot + 255) / 256, 256, 0, stream>>>(
        (const float*)d_in[tab[i].src], wbf + tab[i].dst,
        tab[i].N, tab[i].K, tab[i].pN, tab[i].pK);
  }

  Params prm{};
  for (int i = 0; i < n_in && i < 96; ++i) prm.p[i] = (const float*)d_in[i];
  xlstm_kernel<<<NWG, BLK, 0, stream>>>(prm, ws);

  const float* outs = (const float*)(ws + WS_F32B) + F_OUTS;
  head_kernel<<<128, BLK, 0, stream>>>(outs, wbf + HEADW_BF,
                                       (const float*)d_in[1], (float*)d_out);
}
